// MusicEncoder_39960375722303
// MI455X (gfx1250) — matrix operand order repaired and checked by reading, not compiled, not run on hardware
//
#include <hip/hip_runtime.h>
#include <hip/hip_bf16.h>

// ---------------------------------------------------------------------------
// CDNA5 (gfx1250) fused transformer-VAE encoder.
// Dense math: v_wmma_f32_16x16x32_bf16 (wave32 WMMA).
// Data movement:
//   - GEMM: global_load_async_to_lds_b128 + s_wait_asynccnt, double-buffered
//     64-deep K stages (16 WMMA per barrier interval per wave).
//   - Attention Q/K/V: Tensor Data Mover (tensor_load_to_lds) + s_wait_tensorcnt.
// ---------------------------------------------------------------------------

typedef __bf16 v16bf __attribute__((ext_vector_type(16)));
typedef float  v8f   __attribute__((ext_vector_type(8)));
typedef unsigned int v4u __attribute__((ext_vector_type(4)));
typedef int v8i __attribute__((ext_vector_type(8)));
typedef int v4i __attribute__((ext_vector_type(4)));

// Model constants (from reference)
#define NL   12
#define NH   8
#define DMOD 512
#define DFF  2048
#define DLAT 128
#define SEQ  128
#define BT   8
#define NB   32
#define BBAR (BT * NB)        // 256
#define NTOK (SEQ * BBAR)     // 32768 tokens
#define DLEN 2048

// ---------------------------------------------------------------------------
// WMMA fragment builders (contiguous k order k = 16*half + s, not the ISA table order; wave32)
// A (16x32 bf16): lanes 0-15 row M=lane, K = 0..15; lanes 16-31 same
//                 rows, K = 16..31.
// B (32x16 bf16): lanes 0-15 col N=lane, K = 0..15; lanes 16-31 col N=lane-16,
//                 K = 16..31.
// C/D (16x16 f32): VGPR r, lane l -> M = r + 8*(l>>4), N = l&15.
// Sources are f32 in LDS; reads are explicit float4 (ds_load_b128) and the
// bf16 conversion lowers to v_cvt_pk_bf16_f32.
// ---------------------------------------------------------------------------
__device__ inline void cvt8(v16bf& dst, int o, float4 q0, float4 q1) {
  dst[o + 0] = (__bf16)q0.x; dst[o + 1] = (__bf16)q0.y;
  dst[o + 2] = (__bf16)q0.z; dst[o + 3] = (__bf16)q0.w;
  dst[o + 4] = (__bf16)q1.x; dst[o + 5] = (__bf16)q1.y;
  dst[o + 6] = (__bf16)q1.z; dst[o + 7] = (__bf16)q1.w;
}

__device__ inline v16bf load_a_frag_f32(const float* base, int stride,
                                        int m0, int kbase, int lane) {
  const int m = m0 + (lane & 15);
  const int half = lane >> 4;
  const float* p = base + (size_t)m * stride + kbase;   // 16B-aligned offsets below
  v16bf a;
  cvt8(a, 0, *(const float4*)(p + half * 16),     *(const float4*)(p + half * 16 + 4));
  cvt8(a, 8, *(const float4*)(p + half * 16 + 8), *(const float4*)(p + half * 16 + 12));
  return a;
}

// B source stored [N,K] row-major (e.g. weight W[n][k]).
__device__ inline v16bf load_b_frag_f32(const float* base, int stride,
                                        int n0, int kbase, int lane) {
  const int n = n0 + (lane & 15);
  const int half = lane >> 4;
  const float* p = base + (size_t)n * stride + kbase + half * 16;
  v16bf b;
  cvt8(b, 0, *(const float4*)(p),     *(const float4*)(p + 4));
  cvt8(b, 8, *(const float4*)(p + 8), *(const float4*)(p + 12));
  return b;
}

// B source stored [K,N] row-major (e.g. V[t][d] for P@V); strided reads.
__device__ inline v16bf load_b_frag_f32_tr(const float* base, int stride,
                                           int n0, int kbase, int lane) {
  const int n = n0 + (lane & 15);
  const int half = lane >> 4;
  const float* p = base + (size_t)(kbase + half * 16) * stride + n;
  v16bf b;
#pragma unroll
  for (int j = 0; j < 16; ++j) b[j] = (__bf16)p[j * stride];
  return b;
}

__device__ inline v8f wmma_bf16(v16bf a, v16bf b, v8f c) {
  return __builtin_amdgcn_wmma_f32_16x16x32_bf16(
      /*neg_a=*/false, a, /*neg_b=*/false, b,
      /*c_mod=*/(short)0, c, /*reuse_a=*/false, /*reuse_b=*/false);
}

// Async global->LDS copy of 16 bytes (ASYNCcnt-tracked, CDNA5).
__device__ inline void async_copy_b128(const void* lds_dst, const float* gsrc) {
  const unsigned ldsoff = (unsigned)(unsigned long long)lds_dst;
  asm volatile("global_load_async_to_lds_b128 %0, %1, off"
               :: "v"(ldsoff), "v"(gsrc) : "memory");
}

// ---------------------------------------------------------------------------
// Tensor Data Mover: load a 2D tile (tile_w f32 elements x tile_h rows, row
// stride row_stride_elems) from global into LDS.  D# per CDNA5 ISA §8.3/8.4:
// group0 = {count=1 | lds_addr | global_addr | type=2}, group1 carries
// data_size=4B, tensor dims (= tile dims), tile dims, 48-bit dim0 stride.
// Issue from one wave; TENSORcnt-tracked.
// ---------------------------------------------------------------------------
__device__ inline void tdm_load_2d(void* lds_dst, const float* gsrc,
                                   unsigned tile_w, unsigned tile_h,
                                   unsigned long long row_stride_elems) {
  const unsigned long long ga = (unsigned long long)gsrc;
  v4u g0;
  g0[0] = 1u;                                              // count=1, user mode
  g0[1] = (unsigned)(unsigned long long)lds_dst;           // lds_addr (bytes)
  g0[2] = (unsigned)(ga & 0xFFFFFFFFu);                    // global_addr[31:0]
  g0[3] = (unsigned)((ga >> 32) & 0x1FFFFFFu) | (2u << 30); // ga[56:32] | type=2
  v8i g1;
  g1[0] = (int)(2u << 16);                                 // data_size=4B; no multicast
  g1[1] = (int)((tile_w & 0xFFFFu) << 16);                 // tensor_dim0[15:0]
  g1[2] = (int)(((tile_w >> 16) & 0xFFFFu) |               // tensor_dim0[31:16]
                ((tile_h & 0xFFFFu) << 16));               // tensor_dim1[15:0]
  g1[3] = (int)(((tile_h >> 16) & 0xFFFFu) |               // tensor_dim1[31:16]
                ((tile_w & 0xFFFFu) << 16));               // tile_dim0
  g1[4] = (int)(tile_h & 0xFFFFu);                         // tile_dim1 (tile_dim2=0)
  g1[5] = (int)(unsigned)(row_stride_elems & 0xFFFFFFFFull);   // dim0_stride[31:0]
  g1[6] = (int)(unsigned)((row_stride_elems >> 32) & 0xFFFFull); // dim0_stride[47:32]
  g1[7] = 0;
  v4i z4 = {0, 0, 0, 0};
#if defined(__clang_major__) && (__clang_major__ >= 23)
  v8i z8 = {0, 0, 0, 0, 0, 0, 0, 0};
  __builtin_amdgcn_tensor_load_to_lds(g0, g1, z4, z4, z8, 0);
#else
  __builtin_amdgcn_tensor_load_to_lds(g0, g1, z4, z4, 0);
#endif
}

// ---------------------------------------------------------------------------
// Generic GEMM: C[M,N] = A[M,K] * W[N,K]^T + bias (+relu).
// Block: 256 threads (8 waves). Block tile 128(M) x 128(N); wave w owns the
// 32x64 output at (w>>1, w&1) -> 2x4 accumulators.  K stepped by 64 through
// double-buffered f32 LDS stages (128 KB dynamic LDS) filled with
// global_load_async_to_lds_b128; 16 WMMA per stage per wave.
// Requires: M%128==0, N%128==0, K%64==0 (true for all call sites).
// ---------------------------------------------------------------------------
#define GTM 128
#define GTN 128
#define KS  64

__global__ __launch_bounds__(256) void gemm_xwT_kernel(
    const float* __restrict__ A, const float* __restrict__ W,
    const float* __restrict__ bias, float* __restrict__ C,
    int M, int N, int K, int relu) {
  extern __shared__ float gsm[];
  float* Asf = gsm;                    // 2 x 128*64 f32 (2 x 32 KB)
  float* Bsf = gsm + 2 * GTM * KS;     // 2 x 128*64 f32 (2 x 32 KB)

  const int tid = threadIdx.x;
  const int lane = tid & 31;
  const int w = tid >> 5;
  const int wm = (w >> 1) * 32;        // 0,32,64,96
  const int wn = (w & 1) * 64;         // 0,64
  const int m0 = blockIdx.y * GTM;
  const int n0 = blockIdx.x * GTN;

  // Issue the async fills for one K-stage (16 async instructions per wave).
  auto stage = [&](int buf, int k0) {
    // A tile: 128 rows x 64 floats = 2048 x 16B chunks, 8 per thread.
#pragma unroll
    for (int i = 0; i < 8; ++i) {
      const int c = tid + i * 256;
      const int r = c >> 4, seg = (c & 15) * 4;
      async_copy_b128(&Asf[buf * GTM * KS + r * KS + seg],
                      A + (size_t)(m0 + r) * K + k0 + seg);
    }
    // W tile: 128 rows x 64 floats = 2048 x 16B chunks, 8 per thread.
#pragma unroll
    for (int i = 0; i < 8; ++i) {
      const int c = tid + i * 256;
      const int r = c >> 4, seg = (c & 15) * 4;
      async_copy_b128(&Bsf[buf * GTN * KS + r * KS + seg],
                      W + (size_t)(n0 + r) * K + k0 + seg);
    }
  };

  v8f acc[2][4] = {{{}, {}, {}, {}}, {{}, {}, {}, {}}};

  stage(0, 0);
  int buf = 0;
  for (int k0 = 0; k0 < K; k0 += KS) {
    const bool more = (k0 + KS) < K;
    if (more) stage(buf ^ 1, k0 + KS);
    // Retire the previous stage's 16 asyncs (async loads complete in order);
    // the 16 just issued for the next buffer may remain outstanding.
    if (more) asm volatile("s_wait_asynccnt 0x10" ::: "memory");
    else      asm volatile("s_wait_asynccnt 0x0" ::: "memory");
    __syncthreads();

    const float* At = Asf + buf * GTM * KS;
    const float* Bt = Bsf + buf * GTN * KS;
#pragma unroll
    for (int kk = 0; kk < KS; kk += 32) {
      v16bf a0 = load_a_frag_f32(At, KS, wm, kk, lane);
      v16bf a1 = load_a_frag_f32(At, KS, wm + 16, kk, lane);
#pragma unroll
      for (int jn = 0; jn < 4; ++jn) {
        v16bf bf = load_b_frag_f32(Bt, KS, wn + jn * 16, kk, lane);
        acc[0][jn] = wmma_bf16(a0, bf, acc[0][jn]);
        acc[1][jn] = wmma_bf16(a1, bf, acc[1][jn]);
      }
    }
    __syncthreads();
    buf ^= 1;
  }

  const int half = lane >> 4;
  const int l15 = lane & 15;
#pragma unroll
  for (int im = 0; im < 2; ++im) {
#pragma unroll
    for (int jn = 0; jn < 4; ++jn) {
      const int col = n0 + wn + jn * 16 + l15;
      const float bv = bias ? bias[col] : 0.0f;
#pragma unroll
      for (int r = 0; r < 8; ++r) {
        const int row = m0 + wm + im * 16 + r + 8 * half;
        float v = acc[im][jn][r] + bv;
        if (relu) v = fmaxf(v, 0.0f);
        C[(size_t)row * N + col] = v;
      }
    }
  }
}

// ---------------------------------------------------------------------------
// Fused attention: one block per (bar, head). qkv is [NTOK, 3*DMOD] with
// token = s*BBAR + bar; Q/K/V slices at h*64 (+0/+512/+1024).
// Q/K/V tiles (64 f32 x 128 rows, row stride 256*1536 elems) are loaded by
// the Tensor Data Mover from wave 0; s_wait_tensorcnt 0 + barrier publishes
// them to the block.  Dynamic LDS: Qf/Kf/Vf f32 (3 x 32 KB) + P (64 KB).
// ---------------------------------------------------------------------------
__global__ __launch_bounds__(256) void attn_kernel(
    const float* __restrict__ qkv, const unsigned char* __restrict__ mask,
    float* __restrict__ o) {
  extern __shared__ float asm_f[];
  float* Qf = asm_f;                   // [128][64]
  float* Kf = Qf + SEQ * 64;           // [128][64]
  float* Vf = Kf + SEQ * 64;           // [128][64]  (t-major; consumed transposed)
  float* P  = Vf + SEQ * 64;           // [128][128]

  const int bh = blockIdx.x;           // 0..2047
  const int bar = bh >> 3;
  const int h = bh & 7;
  const int tid = threadIdx.x;
  const int lane = tid & 31;
  const int w = tid >> 5;

  // TDM: wave 0 issues three 2D tile descriptors, waits TENSORcnt, barrier.
  if (tid < 32) {
    const float* qbase = qkv + ((size_t)bar * (3 * DMOD) + h * 64);
    const unsigned long long rstride = (unsigned long long)BBAR * 3 * DMOD;
    tdm_load_2d(Qf, qbase,            64, SEQ, rstride);
    tdm_load_2d(Kf, qbase + DMOD,     64, SEQ, rstride);
    tdm_load_2d(Vf, qbase + 2 * DMOD, 64, SEQ, rstride);
    __builtin_amdgcn_s_wait_tensorcnt((short)0);
  }
  __syncthreads();

  // Scores: P = Q K^T / sqrt(64) + mask_bias.  64 subtiles, 8 per wave.
  const unsigned char* mrow = mask + (size_t)bar * SEQ;
  for (int st = w; st < 64; st += 8) {
    const int tm = (st >> 3) << 4;
    const int tn = (st & 7) << 4;
    v8f acc = {};
#pragma unroll
    for (int ks = 0; ks < 2; ++ks) {
      v16bf af = load_a_frag_f32(Qf, 64, tm, ks * 32, lane);
      v16bf bf = load_b_frag_f32(Kf, 64, tn, ks * 32, lane);  // K is [t][d] = [N][K]
      acc = wmma_bf16(af, bf, acc);
    }
    const int half = lane >> 4;
    const int col = tn + (lane & 15);
    const float mbias = mrow[col] ? -1e9f : 0.0f;
#pragma unroll
    for (int r = 0; r < 8; ++r)
      P[(tm + r + 8 * half) * SEQ + col] = acc[r] * 0.125f + mbias;
  }
  __syncthreads();

  // Row softmax (128 rows; threads 0..127 each own one row).
  if (tid < SEQ) {
    float* row = P + tid * SEQ;
    float mx = -3.4e38f;
    for (int j = 0; j < SEQ; ++j) mx = fmaxf(mx, row[j]);
    float sum = 0.0f;
    for (int j = 0; j < SEQ; ++j) { float e = __expf(row[j] - mx); row[j] = e; sum += e; }
    const float inv = 1.0f / sum;
    for (int j = 0; j < SEQ; ++j) row[j] *= inv;
  }
  __syncthreads();

  // O = P V.  Output tile 128x64: 32 subtiles, 4 per wave.
  for (int st = w; st < 32; st += 8) {
    const int tm = (st >> 2) << 4;
    const int tn = (st & 3) << 4;
    v8f acc = {};
#pragma unroll
    for (int ks = 0; ks < 4; ++ks) {
      v16bf af = load_a_frag_f32(P, SEQ, tm, ks * 32, lane);       // bf16-cast probs
      v16bf bf = load_b_frag_f32_tr(Vf, 64, tn, ks * 32, lane);    // V is [K][N]
      acc = wmma_bf16(af, bf, acc);
    }
    const int half = lane >> 4;
    const int d = tn + (lane & 15);
#pragma unroll
    for (int r = 0; r < 8; ++r) {
      const int s = tm + r + 8 * half;
      o[((size_t)(s * BBAR + bar)) * DMOD + h * 64 + d] = acc[r];
    }
  }
}

// ---------------------------------------------------------------------------
// Token embedding * sqrt(D) + sinusoidal PE.  One block per token.
// ---------------------------------------------------------------------------
__global__ __launch_bounds__(256) void embed_kernel(
    const float* __restrict__ emb, const int* __restrict__ enc_inp,
    float* __restrict__ x) {
  const int token = blockIdx.x;            // s*256 + (bt*32+nb) — same flat order
  const int s = token >> 8;
  const int tok = enc_inp[token];
  const float scale = 22.627416997969522f; // sqrt(512)
  for (int d = threadIdx.x; d < DMOD; d += 256) {
    const int pair = d >> 1;
    const float div = __expf((float)(2 * pair) * (-9.210340371976184f / 512.0f));
    const float ang = (float)s * div;
    const float pe = (d & 1) ? __cosf(ang) : __sinf(ang);
    x[(size_t)token * DMOD + d] = emb[(size_t)tok * DMOD + d] * scale + pe;
  }
}

// ---------------------------------------------------------------------------
// Post-norm LayerNorm, residual fused: x = LN(x + add) * g + b.  D=512.
// One block (256 threads) per row; 2 elements per thread.
// ---------------------------------------------------------------------------
__global__ __launch_bounds__(256) void ln512_kernel(
    float* __restrict__ x, const float* __restrict__ add,
    const float* __restrict__ g, const float* __restrict__ b) {
  __shared__ float red[256];
  const int row = blockIdx.x, tid = threadIdx.x;
  const size_t base = (size_t)row * DMOD;
  const float v0 = x[base + tid] + add[base + tid];
  const float v1 = x[base + 256 + tid] + add[base + 256 + tid];
  red[tid] = v0 + v1;
  __syncthreads();
  for (int s = 128; s > 0; s >>= 1) { if (tid < s) red[tid] += red[tid + s]; __syncthreads(); }
  const float mean = red[0] * (1.0f / 512.0f);
  __syncthreads();
  const float d0 = v0 - mean, d1 = v1 - mean;
  red[tid] = d0 * d0 + d1 * d1;
  __syncthreads();
  for (int s = 128; s > 0; s >>= 1) { if (tid < s) red[tid] += red[tid + s]; __syncthreads(); }
  const float rstd = rsqrtf(red[0] * (1.0f / 512.0f) + 1e-5f);
  x[base + tid]       = d0 * rstd * g[tid]       + b[tid];
  x[base + 256 + tid] = d1 * rstd * g[tid + 256] + b[tid + 256];
}

// latent = mu + eps * exp(0.5*logvar)        (256 x 128)
__global__ __launch_bounds__(256) void latent_kernel(
    const float* __restrict__ mu, const float* __restrict__ lv,
    const float* __restrict__ eps, float* __restrict__ latent) {
  const int i = blockIdx.x * 256 + threadIdx.x;
  latent[i] = mu[i] + eps[i] * __expf(0.5f * lv[i]);
}

// Ragged scatter of bar latents + class embeddings -> [dlen, bt, 192].
__global__ __launch_bounds__(192) void dec_kernel(
    const float* __restrict__ latent, const int* __restrict__ bar_pos,
    const int* __restrict__ rcls, const int* __restrict__ pcls,
    const float* __restrict__ rfw, const float* __restrict__ ppw,
    float* __restrict__ out) {
  const int tb = blockIdx.x;               // t*8 + b
  const int t = tb >> 3, b = tb & 7;
  const int c = threadIdx.x;
  const int* pos = bar_pos + b * (NB + 1);
  int cnt = 0;
  for (int j = 0; j < NB + 1; ++j) cnt += (pos[j] <= t) ? 1 : 0;  // searchsorted right
  const int idx = cnt - 1;
  const bool valid = (idx >= 0) && (idx < NB);
  float v;
  if (c < DLAT) {
    const int ci = valid ? idx : 0;
    v = valid ? latent[((size_t)b * NB + ci) * DLAT + c] : 0.0f;
  } else if (c < DLAT + 32) {
    v = rfw[rcls[tb] * 32 + (c - DLAT)] * 5.656854249492381f;       // sqrt(32)
  } else {
    v = ppw[pcls[tb] * 32 + (c - DLAT - 32)] * 5.656854249492381f;
  }
  out[(size_t)tb * (DLAT + 64) + c] = v;
}

// ---------------------------------------------------------------------------
// Host-side orchestration.
// ---------------------------------------------------------------------------
extern "C" void kernel_launch(void* const* d_in, const int* in_sizes, int n_in,
                              void* d_out, int out_size, void* d_ws, size_t ws_size,
                              hipStream_t stream) {
  const float* tok_emb_w  = (const float*)d_in[0];
  const float* in_proj_w  = (const float*)d_in[1];
  const float* in_proj_b  = (const float*)d_in[2];
  const float* out_proj_w = (const float*)d_in[3];
  const float* out_proj_b = (const float*)d_in[4];
  const float* ff1_w      = (const float*)d_in[5];
  const float* ff1_b      = (const float*)d_in[6];
  const float* ff2_w      = (const float*)d_in[7];
  const float* ff2_b      = (const float*)d_in[8];
  const float* ln1_g      = (const float*)d_in[9];
  const float* ln1_b      = (const float*)d_in[10];
  const float* ln2_g      = (const float*)d_in[11];
  const float* ln2_b      = (const float*)d_in[12];
  const float* fc_mu_w    = (const float*)d_in[13];
  const float* fc_mu_b    = (const float*)d_in[14];
  const float* fc_lv_w    = (const float*)d_in[15];
  const float* fc_lv_b    = (const float*)d_in[16];
  const float* rfreq_emb  = (const float*)d_in[17];
  const float* polyph_emb = (const float*)d_in[18];
  const float* eps        = (const float*)d_in[19];
  const int*   enc_inp    = (const int*)d_in[20];
  // d_in[21] = dec_inp (only its shape matters; unused numerically)
  const int*   bar_pos    = (const int*)d_in[22];
  const int*   rcls       = (const int*)d_in[23];
  const int*   pcls       = (const int*)d_in[24];
  const unsigned char* mask = (const unsigned char*)d_in[25];

  float* out = (float*)d_out;              // [mu | logvar | dec_seg_emb_cat]
  // Workspace layout (floats):
  //   x    : 32768 x 512   = 16,777,216
  //   hbuf : max(qkv 32768x1536, ffn 32768x2048) = 67,108,864 (reused)
  //   obuf : 32768 x 512   = 16,777,216
  //   latent: 256 x 128
  float* x      = (float*)d_ws;
  float* hbuf   = x + (size_t)NTOK * DMOD;
  float* obuf   = hbuf + (size_t)NTOK * DFF;
  float* latent = obuf + (size_t)NTOK * DMOD;

  const int GEMM_SMEM = (2 * GTM * KS + 2 * GTN * KS) * (int)sizeof(float);   // 128 KB
  const int ATT_SMEM  = (3 * SEQ * 64 + SEQ * SEQ) * (int)sizeof(float);      // 160 KB
  hipFuncSetAttribute(reinterpret_cast<const void*>(gemm_xwT_kernel),
                      hipFuncAttributeMaxDynamicSharedMemorySize, GEMM_SMEM);
  hipFuncSetAttribute(reinterpret_cast<const void*>(attn_kernel),
                      hipFuncAttributeMaxDynamicSharedMemorySize, ATT_SMEM);

  embed_kernel<<<NTOK, 256, 0, stream>>>(tok_emb_w, enc_inp, x);

  for (int i = 0; i < NL; ++i) {
    // qkv = x @ in_proj_w^T + b       [32768, 1536]
    gemm_xwT_kernel<<<dim3(3 * DMOD / GTN, NTOK / GTM), 256, GEMM_SMEM, stream>>>(
        x, in_proj_w + (size_t)i * 3 * DMOD * DMOD, in_proj_b + (size_t)i * 3 * DMOD,
        hbuf, NTOK, 3 * DMOD, DMOD, 0);
    // attention -> obuf [32768, 512]
    attn_kernel<<<BBAR * NH, 256, ATT_SMEM, stream>>>(hbuf, mask, obuf);
    // o @ out_proj^T + b -> hbuf (qkv is dead)
    gemm_xwT_kernel<<<dim3(DMOD / GTN, NTOK / GTM), 256, GEMM_SMEM, stream>>>(
        obuf, out_proj_w + (size_t)i * DMOD * DMOD, out_proj_b + (size_t)i * DMOD,
        hbuf, NTOK, DMOD, DMOD, 0);
    // x = LN(x + attn_out)
    ln512_kernel<<<NTOK, 256, 0, stream>>>(x, hbuf, ln1_g + (size_t)i * DMOD,
                                           ln1_b + (size_t)i * DMOD);
    // h = relu(x @ ff1^T + b)         [32768, 2048]
    gemm_xwT_kernel<<<dim3(DFF / GTN, NTOK / GTM), 256, GEMM_SMEM, stream>>>(
        x, ff1_w + (size_t)i * DFF * DMOD, ff1_b + (size_t)i * DFF,
        hbuf, NTOK, DFF, DMOD, 1);
    // y = h @ ff2^T + b -> obuf
    gemm_xwT_kernel<<<dim3(DMOD / GTN, NTOK / GTM), 256, GEMM_SMEM, stream>>>(
        hbuf, ff2_w + (size_t)i * DMOD * DFF, ff2_b + (size_t)i * DMOD,
        obuf, NTOK, DMOD, DFF, 0);
    // x = LN(x + y)
    ln512_kernel<<<NTOK, 256, 0, stream>>>(x, obuf, ln2_g + (size_t)i * DMOD,
                                           ln2_b + (size_t)i * DMOD);
  }

  // hidden = x[0] == first 256 rows (tokens s=0).  mu / logvar into d_out.
  gemm_xwT_kernel<<<dim3(DLAT / GTN, BBAR / GTM), 256, GEMM_SMEM, stream>>>(
      x, fc_mu_w, fc_mu_b, out, BBAR, DLAT, DMOD, 0);
  gemm_xwT_kernel<<<dim3(DLAT / GTN, BBAR / GTM), 256, GEMM_SMEM, stream>>>(
      x, fc_lv_w, fc_lv_b, out + BBAR * DLAT, BBAR, DLAT, DMOD, 0);

  latent_kernel<<<BBAR * DLAT / 256, 256, 0, stream>>>(out, out + BBAR * DLAT, eps, latent);

  dec_kernel<<<DLEN * BT, DLAT + 64, 0, stream>>>(
      latent, bar_pos, rcls, pcls, rfreq_emb, polyph_emb, out + 2 * BBAR * DLAT);
}
